// Attention_1855425872447
// MI455X (gfx1250) — matrix operand order repaired and checked by reading, not compiled, not run on hardware
//
#include <hip/hip_runtime.h>

typedef __attribute__((ext_vector_type(16))) _Float16 v16h;
typedef __attribute__((ext_vector_type(8)))  _Float16 v8h;
typedef __attribute__((ext_vector_type(4)))  _Float16 v4h;
typedef __attribute__((ext_vector_type(8)))  float    v8f;
typedef __attribute__((ext_vector_type(4)))  int      v4i;

constexpr int B = 8, N = 1024, C = 768, H = 12, HD = 64;
constexpr int C3 = 3 * C;
constexpr float SCALE = 0.125f;   // HD^-0.5

// Padded LDS row strides (halves): conflict-free 16-lane fragment access on
// 64x4B banks, still 16B-aligned for ds b128.
constexpr int WSTR = 40;   // weight tile rows (80B)
constexpr int KSTR = 72;   // K tile rows (144B)
constexpr int VSTR = 40;   // V^T tile rows (80B)
constexpr int PSTR = 40;   // P tile rows (80B)

#if __has_builtin(__builtin_amdgcn_global_load_async_to_lds_b128)
#define HAVE_ASYNC_LDS 1
#else
#define HAVE_ASYNC_LDS 0
#endif

// Builtin signature: (v4i addrspace(1)*, v4i addrspace(3)*, imm offset, imm cpol).
#define AS1V4I(p) ((__attribute__((address_space(1))) v4i*)(unsigned long long)(p))
#define AS3V4I(p) ((__attribute__((address_space(3))) v4i*)(unsigned)(unsigned long long)(p))

__device__ __forceinline__ void async_copy_32B(const _Float16* src, _Float16* dst) {
#if HAVE_ASYNC_LDS
  __builtin_amdgcn_global_load_async_to_lds_b128(AS1V4I(src), AS3V4I(dst), 0, 0);
  __builtin_amdgcn_global_load_async_to_lds_b128(AS1V4I(src + 8), AS3V4I(dst + 8), 0, 0);
#else
  *(v8h*)dst = *(const v8h*)src;
  *(v8h*)(dst + 8) = *(const v8h*)(src + 8);
#endif
}

__device__ __forceinline__ void async_commit_barrier() {
#if HAVE_ASYNC_LDS
#if __has_builtin(__builtin_amdgcn_s_wait_asynccnt)
  __builtin_amdgcn_s_wait_asynccnt(0);
#else
  asm volatile("s_wait_asynccnt 0x0" ::: "memory");
#endif
#endif
  __syncthreads();
}

__device__ __forceinline__ v8f wmma_f16(v16h a, v16h b, v8f c) {
  return __builtin_amdgcn_wmma_f32_16x16x32_f16(false, a, false, b, (short)0, c, false, false);
}

// A-fragment (16x32 f16): 16 contiguous halves (two 8-half chunks at +0 and +8).
__device__ __forceinline__ v16h load_afrag(const _Float16* p) {
  v8h lo = *(const v8h*)p;
  v8h hh = *(const v8h*)(p + 8);
  return __builtin_shufflevector(lo, hh, 0, 1, 2, 3, 4, 5, 6, 7,
                                 8, 9, 10, 11, 12, 13, 14, 15);
}
// B-fragment from LDS: 16 contiguous halves, 16B-aligned (two ds b128).
__device__ __forceinline__ v16h load_bfrag_lds(const _Float16* p) {
  v8h lo = *(const v8h*)p;
  v8h hh = *(const v8h*)(p + 8);
  return __builtin_shufflevector(lo, hh, 0, 1, 2, 3, 4, 5, 6, 7,
                                 8, 9, 10, 11, 12, 13, 14, 15);
}

// ---------------------------------------------------------------------------
// f32 -> f16 bulk convert (vectorized x4)
// ---------------------------------------------------------------------------
__global__ void __launch_bounds__(256) cvt_f16_kernel(
    const float* __restrict__ src, _Float16* __restrict__ dst, int n4)
{
  const int i = blockIdx.x * 256 + threadIdx.x;
  if (i >= n4) return;
  const float4 f = ((const float4*)src)[i];
  v4h h;
  h[0] = (_Float16)f.x; h[1] = (_Float16)f.y;
  h[2] = (_Float16)f.z; h[3] = (_Float16)f.w;
  ((v4h*)dst)[i] = h;
}

// Stage one 64(col) x 32(k) f16 weight tile into padded LDS, 32B per thread.
__device__ __forceinline__ void stage_wtile(const _Float16* w, int ntile0, int kb,
                                            _Float16* buf, int tid) {
  const int c = tid >> 1;              // 0..63
  const int half = (tid & 1) * 16;     // 0 or 16
  async_copy_32B(w + (size_t)(ntile0 + c) * C + kb + half, buf + c * WSTR + half);
}

// ---------------------------------------------------------------------------
// Kernel 1: qkv = x @ Wqkv^T + bias -> q (scaled), k, v in [B,H,N,HD] f16
// Weight tiles async-staged to LDS, double-buffered; A pipelined in registers.
// ---------------------------------------------------------------------------
__global__ void __launch_bounds__(128) qkv_proj_kernel(
    const _Float16* __restrict__ x16, const _Float16* __restrict__ w16,
    const float* __restrict__ qbias, const float* __restrict__ vbias,
    _Float16* __restrict__ q_ws, _Float16* __restrict__ k_ws,
    _Float16* __restrict__ v_ws)
{
  __shared__ _Float16 wbuf[2][64 * WSTR];
  const int wave = threadIdx.x >> 5;
  const int lane = threadIdx.x & 31;
  const int hi = lane >> 4, l16 = lane & 15;
  const int ntile0 = blockIdx.x * 64;
  const int mtile  = blockIdx.y * 64 + wave * 16;
  const _Float16* xr = x16 + (size_t)(mtile + l16) * C + hi * 16;

  stage_wtile(w16, ntile0, 0, wbuf[0], threadIdx.x);
  v16h a = load_afrag(xr);
  v8f acc[4] = {v8f{}, v8f{}, v8f{}, v8f{}};

  constexpr int KS = C / 32;   // 24
  async_commit_barrier();
  for (int i = 0; i < KS; ++i) {
    const int cur = i & 1;
    if (i + 1 < KS) stage_wtile(w16, ntile0, (i + 1) * 32, wbuf[cur ^ 1], threadIdx.x);
    v16h an = a;
    if (i + 1 < KS) an = load_afrag(xr + (i + 1) * 32);
    const _Float16* wb = wbuf[cur];
    acc[0] = wmma_f16(a, load_bfrag_lds(wb + (l16)      * WSTR + hi * 16), acc[0]);
    acc[1] = wmma_f16(a, load_bfrag_lds(wb + (16 + l16) * WSTR + hi * 16), acc[1]);
    acc[2] = wmma_f16(a, load_bfrag_lds(wb + (32 + l16) * WSTR + hi * 16), acc[2]);
    acc[3] = wmma_f16(a, load_bfrag_lds(wb + (48 + l16) * WSTR + hi * 16), acc[3]);
    a = an;
    async_commit_barrier();
  }

  // one 64-wide tile always lies inside a single q/k/v section (768 % 64 == 0)
  const int sec = ntile0 / C;
  const int c0  = ntile0 - sec * C;
  _Float16* dst = (sec == 0) ? q_ws : ((sec == 1) ? k_ws : v_ws);
  const float mul = (sec == 0) ? SCALE : 1.f;
#pragma unroll
  for (int ct = 0; ct < 4; ++ct) {
    const int c = c0 + ct * 16 + l16;
    const int hh = c >> 6, hd = c & 63;
    const float bias = (sec == 0) ? qbias[c] : ((sec == 2) ? vbias[c] : 0.f);
#pragma unroll
    for (int r = 0; r < 8; ++r) {
      const int g = mtile + r + hi * 8;
      dst[(((size_t)((g >> 10) * H + hh)) * N + (g & 1023)) * HD + hd] =
          (_Float16)((acc[ct][r] + bias) * mul);
    }
  }
}

// ---------------------------------------------------------------------------
// Kernel 2: flash attention per (b, h, 64-query tile); 4 waves/block.
// Double-buffered K/V staging: async K copies + transposed V stores for chunk
// i+1 issue right after the commit barrier, overlapping chunk i's compute.
// ---------------------------------------------------------------------------
__global__ void __launch_bounds__(128) flash_attn_kernel(
    const _Float16* __restrict__ q_ws, const _Float16* __restrict__ k_ws,
    const _Float16* __restrict__ v_ws, const float* __restrict__ rpb,
    const int* __restrict__ mask, _Float16* __restrict__ o_ws)
{
  __shared__ _Float16 kbuf[2][32 * KSTR];   // [key][hd], padded
  __shared__ _Float16 vtbuf[2][64 * VSTR];  // [hd][key], padded
  __shared__ _Float16 pbuf[4][16 * PSTR];   // per-wave P relayout C/D -> A

  const int wg = blockIdx.x;
  const int qt = wg & 15;
  const int h  = (wg >> 4) % H;
  const int b  = wg / (16 * H);
  const int wave = threadIdx.x >> 5;
  const int lane = threadIdx.x & 31;
  const int hi = lane >> 4, l16 = lane & 15;

  const size_t bh = (size_t)(b * H + h);
  const int m0 = qt * 64 + wave * 16;

  const _Float16* qp = q_ws + (bh * N + m0 + l16) * HD + hi * 16;
  const v16h qf0 = load_afrag(qp);
  const v16h qf1 = load_afrag(qp + 32);

  v8f oacc[4] = {v8f{}, v8f{}, v8f{}, v8f{}};
  float mi[8], li[8];
#pragma unroll
  for (int r = 0; r < 8; ++r) { mi[r] = -1e30f; li[r] = 0.f; }

  const _Float16* kp = k_ws + bh * (size_t)N * HD;
  const _Float16* vp = v_ws + bh * (size_t)N * HD;
  const float* rp0 = rpb + ((size_t)h * N + m0 + hi * 8) * N;
  const int* mrow = mask + b * N;

  const int key = threadIdx.x >> 2;           // staging role: 32B per thread
  const int seg = (threadIdx.x & 3) * 16;

  auto stage = [&](int buf, int kc) {
    const _Float16* krow = kp + (size_t)(kc + key) * HD + seg;
    const _Float16* vrow = vp + (size_t)(kc + key) * HD + seg;
    async_copy_32B(krow, &kbuf[buf][key * KSTR + seg]);
    const v16h vv = *(const v16h*)vrow;
#pragma unroll
    for (int j = 0; j < 16; ++j) vtbuf[buf][(seg + j) * VSTR + key] = vv[j];
    if (kc + 32 < N) {
      __builtin_prefetch(krow + 32 * HD, 0, 0);
      __builtin_prefetch(vrow + 32 * HD, 0, 0);
    }
  };

  stage(0, 0);
  constexpr int NC = N / 32;   // 32 chunks
  for (int i = 0; i < NC; ++i) {
    const int kc = i * 32;
    const int cur = i & 1;
    async_commit_barrier();                       // current buffer ready
    if (i + 1 < NC) stage(cur ^ 1, kc + 32);      // overlap next chunk's staging

    // S = Q K^T : two 16-key tiles, K-dim = HD split 32+32
    const _Float16* kb = kbuf[cur];
    v8f s0 = {}, s1 = {};
    s0 = wmma_f16(qf0, load_bfrag_lds(kb + l16 * KSTR + hi * 16), s0);
    s0 = wmma_f16(qf1, load_bfrag_lds(kb + l16 * KSTR + 32 + hi * 16), s0);
    s1 = wmma_f16(qf0, load_bfrag_lds(kb + (16 + l16) * KSTR + hi * 16), s1);
    s1 = wmma_f16(qf1, load_bfrag_lds(kb + (16 + l16) * KSTR + 32 + hi * 16), s1);

    const int msk0 = mrow[kc + l16];
    const int msk1 = mrow[kc + 16 + l16];
    const float* rp = rp0 + kc + l16;
#pragma unroll
    for (int r = 0; r < 8; ++r) {
      float a0 = msk0 ? -1e30f : (s0[r] + rp[(size_t)r * N]);
      float a1 = msk1 ? -1e30f : (s1[r] + rp[(size_t)r * N + 16]);
      float mx = fmaxf(a0, a1);
      mx = fmaxf(mx, __shfl_xor(mx, 1, 32));
      mx = fmaxf(mx, __shfl_xor(mx, 2, 32));
      mx = fmaxf(mx, __shfl_xor(mx, 4, 32));
      mx = fmaxf(mx, __shfl_xor(mx, 8, 32));
      const float mn = fmaxf(mi[r], mx);
      const float sc = __expf(mi[r] - mn);
      const float e0 = __expf(a0 - mn);
      const float e1 = __expf(a1 - mn);
      float rs = e0 + e1;
      rs += __shfl_xor(rs, 1, 32);
      rs += __shfl_xor(rs, 2, 32);
      rs += __shfl_xor(rs, 4, 32);
      rs += __shfl_xor(rs, 8, 32);
      li[r] = li[r] * sc + rs;
      mi[r] = mn;
      oacc[0][r] *= sc; oacc[1][r] *= sc; oacc[2][r] *= sc; oacc[3][r] *= sc;
      pbuf[wave][(r + hi * 8) * PSTR + l16]      = (_Float16)e0;
      pbuf[wave][(r + hi * 8) * PSTR + 16 + l16] = (_Float16)e1;
    }
    __syncthreads();   // P tile visible in A-fragment order

    const v16h pf = load_afrag(&pbuf[wave][l16 * PSTR + hi * 16]);
#pragma unroll
    for (int c = 0; c < 4; ++c)
      oacc[c] = wmma_f16(pf, load_bfrag_lds(&vtbuf[cur][(c * 16 + l16) * VSTR + hi * 16]),
                         oacc[c]);
  }

  _Float16* op = o_ws + ((size_t)b * N + m0) * C + h * HD;
#pragma unroll
  for (int r = 0; r < 8; ++r) {
    const float inv = 1.f / li[r];
#pragma unroll
    for (int c = 0; c < 4; ++c)
      op[(size_t)(r + hi * 8) * C + c * 16 + l16] = (_Float16)(oacc[c][r] * inv);
  }
}

// ---------------------------------------------------------------------------
// Kernel 3: out = attn_out @ proj_weight^T + proj_bias (f32 out).
// Same async double-buffered LDS weight staging as kernel 1.
// ---------------------------------------------------------------------------
__global__ void __launch_bounds__(128) out_proj_kernel(
    const _Float16* __restrict__ o_ws, const _Float16* __restrict__ pw16,
    const float* __restrict__ pbias, float* __restrict__ out)
{
  __shared__ _Float16 wbuf[2][64 * WSTR];
  const int wave = threadIdx.x >> 5;
  const int lane = threadIdx.x & 31;
  const int hi = lane >> 4, l16 = lane & 15;
  const int ntile0 = blockIdx.x * 64;
  const int mtile  = blockIdx.y * 64 + wave * 16;
  const _Float16* ar = o_ws + (size_t)(mtile + l16) * C + hi * 16;

  stage_wtile(pw16, ntile0, 0, wbuf[0], threadIdx.x);
  v16h a = load_afrag(ar);
  v8f acc[4] = {v8f{}, v8f{}, v8f{}, v8f{}};

  constexpr int KS = C / 32;
  async_commit_barrier();
  for (int i = 0; i < KS; ++i) {
    const int cur = i & 1;
    if (i + 1 < KS) stage_wtile(pw16, ntile0, (i + 1) * 32, wbuf[cur ^ 1], threadIdx.x);
    v16h an = a;
    if (i + 1 < KS) an = load_afrag(ar + (i + 1) * 32);
    const _Float16* wb = wbuf[cur];
    acc[0] = wmma_f16(a, load_bfrag_lds(wb + (l16)      * WSTR + hi * 16), acc[0]);
    acc[1] = wmma_f16(a, load_bfrag_lds(wb + (16 + l16) * WSTR + hi * 16), acc[1]);
    acc[2] = wmma_f16(a, load_bfrag_lds(wb + (32 + l16) * WSTR + hi * 16), acc[2]);
    acc[3] = wmma_f16(a, load_bfrag_lds(wb + (48 + l16) * WSTR + hi * 16), acc[3]);
    a = an;
    async_commit_barrier();
  }
#pragma unroll
  for (int ct = 0; ct < 4; ++ct) {
    const int j = ntile0 + ct * 16 + l16;
    const float bias = pbias[j];
#pragma unroll
    for (int r = 0; r < 8; ++r)
      out[(size_t)(mtile + r + hi * 8) * C + j] = acc[ct][r] + bias;
  }
}

// ---------------------------------------------------------------------------
extern "C" void kernel_launch(void* const* d_in, const int* in_sizes, int n_in,
                              void* d_out, int out_size, void* d_ws, size_t ws_size,
                              hipStream_t stream)
{
  const float* x    = (const float*)d_in[0];
  const int*   mask = (const int*)d_in[1];
  const float* rpb  = (const float*)d_in[2];
  const float* wqkv = (const float*)d_in[3];
  const float* qb   = (const float*)d_in[4];
  const float* vb   = (const float*)d_in[5];
  const float* pw   = (const float*)d_in[6];
  const float* pb   = (const float*)d_in[7];
  float* out = (float*)d_out;

  const size_t bnc = (size_t)B * N * C;           // 6,291,456 (== B*H*N*HD)
  _Float16* q_ws  = (_Float16*)d_ws;
  _Float16* k_ws  = q_ws + bnc;
  _Float16* v_ws  = k_ws + bnc;
  _Float16* o_ws  = v_ws + bnc;
  _Float16* x16   = o_ws + bnc;
  _Float16* w16   = x16 + bnc;                    // 3C x C = 1,769,472
  _Float16* pw16  = w16 + (size_t)C3 * C;         // C x C  =   589,824

  const int nx4 = (int)(bnc / 4);
  const int nw4 = (C3 * C) / 4;
  const int np4 = (C * C) / 4;
  cvt_f16_kernel<<<(nx4 + 255) / 256, 256, 0, stream>>>(x, x16, nx4);
  cvt_f16_kernel<<<(nw4 + 255) / 256, 256, 0, stream>>>(wqkv, w16, nw4);
  cvt_f16_kernel<<<(np4 + 255) / 256, 256, 0, stream>>>(pw, pw16, np4);

  qkv_proj_kernel<<<dim3(C3 / 64, (B * N) / 64), 128, 0, stream>>>(
      x16, w16, qb, vb, q_ws, k_ws, v_ws);
  flash_attn_kernel<<<dim3(B * H * (N / 64)), 128, 0, stream>>>(
      q_ws, k_ws, v_ws, rpb, mask, o_ws);
  out_proj_kernel<<<dim3(C / 64, (B * N) / 64), 128, 0, stream>>>(
      o_ws, pw16, pb, out);

  (void)in_sizes; (void)n_in; (void)out_size; (void)ws_size;
}
